// ReplicateMaskAttention_25503515804228
// MI455X (gfx1250) — matrix operand order repaired and checked by reading, not compiled, not run on hardware
//
#include <hip/hip_runtime.h>

typedef __attribute__((ext_vector_type(16))) _Float16 v16h;
typedef __attribute__((ext_vector_type(8)))  float    v8f;

#define B_  64
#define T_  512
#define R_  4
#define D_  256
#define TPAD 264          // padded row stride (halves) for 256-wide f16 tiles
#define WST  40           // padded row stride (halves) for 32-wide W chunks
#define GROUPS_PER_WG 16  // (b,t) groups per workgroup
#define ROWS_PER_WG   64
#define NEGINF (-1000000000.0f)

struct alignas(16) Smem {
  _Float16 XA[ROWS_PER_WG * TPAD]; // X tile; reused for attended tile
  _Float16 Qh[ROWS_PER_WG * TPAD];
  _Float16 Kh[ROWS_PER_WG * TPAD];
  _Float16 Vh[ROWS_PER_WG * TPAD];
  _Float16 Wc[256 * WST];          // current W K-chunk: 256 n-rows x 32 k (padded)
  float    bias[4 * 256];          // bq | bk | bv | bo
  float    valid[ROWS_PER_WG];     // per-row replicate mask as 0/1 float
};                                 // 160000 bytes total

union ABv { v16h v; float4 q[2]; };
union H16 { float4 q[2]; _Float16 h[16]; };

// ---------------------------------------------------------------------------
// Prep: fp32 weights -> f16, tiled [proj][kChunk(8)][n(256)][k(32)] for
// coalesced 64B-per-row streaming in the main kernel.
// ---------------------------------------------------------------------------
__global__ void prep_weights_f16(const float* __restrict__ Wq, const float* __restrict__ Wk,
                                 const float* __restrict__ Wv, const float* __restrict__ Wo,
                                 _Float16* __restrict__ Wh)
{
  int tid = blockIdx.x * 256 + threadIdx.x;      // 4*256*256 = 262144 threads
  int p = tid >> 16; int rem = tid & 0xFFFF;
  int n = rem >> 8;  int k = rem & 0xFF;
  const float* W = (p == 0) ? Wq : (p == 1) ? Wk : (p == 2) ? Wv : Wo;
  float w = W[n * 256 + k];
  int kc = k >> 5, kh = k & 31;
  Wh[((p * 8 + kc) * 256 + n) * 32 + kh] = (_Float16)w;
}

// ---------------------------------------------------------------------------
// 64x256 GEMM tile: out = A(64x256,f16 LDS) * W^T (W streamed f16 from L2).
// 8 waves: wave = (mBlock 0..3) x (nBlock 0..1); each wave 16 rows x 128 cols
// = 8 accumulator tiles of v_wmma_f32_16x16x32_f16.
// ---------------------------------------------------------------------------
__device__ inline void gemm_64x256(Smem& s, const _Float16* __restrict__ Alds,
                                   const _Float16* __restrict__ Whp,
                                   int tid, int lane, int mBase, int nBase, v8f acc[8])
{
  v8f zero = {};
#pragma unroll
  for (int nt = 0; nt < 8; nt++) acc[nt] = zero;
  const int l15  = lane & 15;
  const int aOff = (lane >> 4) * 8;   // A: hi half-wave holds K+8 (halves)
  const int bOff = (lane >> 4) * 8;   // B: hi half-wave holds K+8 (halves)
  const _Float16* arow = Alds + (mBase + l15) * TPAD;
  for (int kc = 0; kc < 8; kc++) {
    { // stage W chunk: thread t copies n-row t (32 halves = 64B), coalesced
      const float4* src = (const float4*)(Whp + (kc * 256 + tid) * 32);
      float4* dst = (float4*)&s.Wc[tid * WST];
      dst[0] = src[0]; dst[1] = src[1]; dst[2] = src[2]; dst[3] = src[3];
    }
    __syncthreads();
    ABv a;  // 16-bit A 16x32 layout: VGPR0-3 K{0..7}+aOff, VGPR4-7 K{16..23}+aOff
    a.q[0] = *(const float4*)(arow + kc * 32 + aOff);
    a.q[1] = *(const float4*)(arow + kc * 32 + 16 + aOff);
#pragma unroll
    for (int nt = 0; nt < 8; nt++) {
      const _Float16* brow = s.Wc + (nBase + nt * 16 + l15) * WST;
      ABv b;  // B: lane = column n, K{0..7}+bOff then K{16..23}+bOff (same as A)
      b.q[0] = *(const float4*)(brow + bOff);
      b.q[1] = *(const float4*)(brow + bOff + 16);
      acc[nt] = __builtin_amdgcn_wmma_f32_16x16x32_f16(
          false, a.v, false, b.v, (short)0, acc[nt], false, false);
    }
    __syncthreads();
  }
}

__device__ inline void store_proj(Smem& s, _Float16* __restrict__ dst, const v8f acc[8],
                                  int lane, int mBase, int nBase, int biasBase)
{
  const int l15 = lane & 15, hi8 = (lane >> 4) * 8;
#pragma unroll
  for (int nt = 0; nt < 8; nt++) {
    int n = nBase + nt * 16 + l15;
    float bb = s.bias[biasBase + n];
#pragma unroll
    for (int j = 0; j < 8; j++) {
      int m = mBase + j + hi8;   // C layout: VGPR j -> row j (+8 for hi half)
      dst[m * TPAD + n] = (_Float16)(acc[nt][j] + bb);
    }
  }
}

// ---------------------------------------------------------------------------
// Fused: QKV projections -> masked 4x4 softmax attention -> O projection.
// ---------------------------------------------------------------------------
__global__ __launch_bounds__(256) void fused_rep_attn(
    const float* __restrict__ x, const int* __restrict__ rmask,
    const _Float16* __restrict__ Wh,
    const float* __restrict__ bq, const float* __restrict__ bk,
    const float* __restrict__ bv, const float* __restrict__ bo,
    float* __restrict__ out)
{
  __shared__ Smem s;
  const int tid  = threadIdx.x;
  const int lane = tid & 31;
  const int wave = tid >> 5;
  const int mBase = (wave & 3) * 16;
  const int nBase = (wave >> 2) * 128;
  const int row0  = blockIdx.x * ROWS_PER_WG;   // rows are (b*T+t)*R + r, contiguous

  // biases + per-row validity -> LDS
  s.bias[tid]       = bq[tid];
  s.bias[256 + tid] = bk[tid];
  s.bias[512 + tid] = bv[tid];
  s.bias[768 + tid] = bo[tid];
  if (tid < ROWS_PER_WG) {
    int gr = row0 + tid;                        // b = gr>>11 (T*R=2048), r = gr&3
    s.valid[tid] = (rmask[((gr >> 11) << 2) | (gr & 3)] != 0) ? 1.0f : 0.0f;
  }

  // X tile (64 rows x 256 f32) -> f16 LDS, coalesced float4 loads
  const float* xt = x + (size_t)row0 * D_;
  for (int i = 0; i < 16; i++) {
    int idx = i * 1024 + tid * 4;
    float4 v = *(const float4*)(xt + idx);
    _Float16* d = &s.XA[(idx >> 8) * TPAD + (idx & 255)];
    d[0] = (_Float16)v.x; d[1] = (_Float16)v.y; d[2] = (_Float16)v.z; d[3] = (_Float16)v.w;
  }
  __syncthreads();

  v8f acc[8];
  gemm_64x256(s, s.XA, Wh + 0 * 65536, tid, lane, mBase, nBase, acc);
  store_proj(s, s.Qh, acc, lane, mBase, nBase, 0);
  gemm_64x256(s, s.XA, Wh + 1 * 65536, tid, lane, mBase, nBase, acc);
  store_proj(s, s.Kh, acc, lane, mBase, nBase, 256);
  gemm_64x256(s, s.XA, Wh + 2 * 65536, tid, lane, mBase, nBase, acc);
  store_proj(s, s.Vh, acc, lane, mBase, nBase, 512);
  __syncthreads();

  // ---- attention: 16 threads per (b,t) group, each owns a 16-wide d-chunk ----
  {
    const int g = tid >> 4;
    const int c = tid & 15;
    const int dOff = c * 16;
    H16 qh[4], kh[4], vh[4];
#pragma unroll
    for (int r = 0; r < 4; r++) {
      const _Float16* qp = &s.Qh[(g * 4 + r) * TPAD + dOff];
      qh[r].q[0] = *(const float4*)(qp); qh[r].q[1] = *(const float4*)(qp + 8);
      const _Float16* kp = &s.Kh[(g * 4 + r) * TPAD + dOff];
      kh[r].q[0] = *(const float4*)(kp); kh[r].q[1] = *(const float4*)(kp + 8);
      const _Float16* vp = &s.Vh[(g * 4 + r) * TPAD + dOff];
      vh[r].q[0] = *(const float4*)(vp); vh[r].q[1] = *(const float4*)(vp + 8);
    }
    float p[4][4] = {};
#pragma unroll
    for (int dd = 0; dd < 16; dd++) {
      float qf[4], kf[4];
#pragma unroll
      for (int r = 0; r < 4; r++) { qf[r] = (float)qh[r].h[dd]; kf[r] = (float)kh[r].h[dd]; }
#pragma unroll
      for (int r = 0; r < 4; r++)
#pragma unroll
        for (int ss = 0; ss < 4; ss++) p[r][ss] += qf[r] * kf[ss];
    }
    // reduce partial dots across the 16 lanes of this group (wave32 halves)
#pragma unroll
    for (int r = 0; r < 4; r++)
#pragma unroll
      for (int ss = 0; ss < 4; ss++)
#pragma unroll
        for (int o = 1; o < 16; o <<= 1)
          p[r][ss] += __shfl_xor(p[r][ss], o, 32);

    float kvf[4], attn[4][4];
#pragma unroll
    for (int ss = 0; ss < 4; ss++) kvf[ss] = s.valid[g * 4 + ss];
#pragma unroll
    for (int r = 0; r < 4; r++) {
      float qv = s.valid[g * 4 + r];
      float sc[4], mx = NEGINF;
#pragma unroll
      for (int ss = 0; ss < 4; ss++) {
        sc[ss] = (qv * kvf[ss] > 0.f) ? p[r][ss] * 0.0625f : NEGINF;  // 1/sqrt(256)
        mx = fmaxf(mx, sc[ss]);
      }
      float sum = 0.f;
#pragma unroll
      for (int ss = 0; ss < 4; ss++) { sc[ss] = __expf(sc[ss] - mx); sum += sc[ss]; }
      float inv = 1.0f / sum;
#pragma unroll
      for (int ss = 0; ss < 4; ss++) attn[r][ss] = sc[ss] * inv * kvf[ss];
    }
    // attended = attn * V, written (f16) into the retired X buffer
#pragma unroll
    for (int r = 0; r < 4; r++) {
      H16 o16;
#pragma unroll
      for (int dd = 0; dd < 16; dd++) {
        float a = 0.f;
#pragma unroll
        for (int ss = 0; ss < 4; ss++) a += attn[r][ss] * (float)vh[ss].h[dd];
        o16.h[dd] = (_Float16)a;
      }
      _Float16* ap = &s.XA[(g * 4 + r) * TPAD + dOff];
      *(float4*)(ap) = o16.q[0]; *(float4*)(ap + 8) = o16.q[1];
    }
  }
  // (gemm's first internal barrier orders attended-writes before A-reads)

  // ---- O projection + bias + query-mask zeroing -> fp32 out ----
  gemm_64x256(s, s.XA, Wh + 3 * 65536, tid, lane, mBase, nBase, acc);
  {
    const int l15 = lane & 15, hi8 = (lane >> 4) * 8;
#pragma unroll
    for (int nt = 0; nt < 8; nt++) {
      int n = nBase + nt * 16 + l15;
      float bb = s.bias[768 + n];
#pragma unroll
      for (int j = 0; j < 8; j++) {
        int m = mBase + j + hi8;
        out[(size_t)(row0 + m) * D_ + n] = (acc[nt][j] + bb) * s.valid[m];
      }
    }
  }
}

extern "C" void kernel_launch(void* const* d_in, const int* in_sizes, int n_in,
                              void* d_out, int out_size, void* d_ws, size_t ws_size,
                              hipStream_t stream)
{
  (void)in_sizes; (void)n_in; (void)out_size; (void)ws_size;
  const float* x  = (const float*)d_in[0];
  const int*   rm = (const int*)d_in[1];
  const float* Wq = (const float*)d_in[2];
  const float* bq = (const float*)d_in[3];
  const float* Wk = (const float*)d_in[4];
  const float* bk = (const float*)d_in[5];
  const float* Wv = (const float*)d_in[6];
  const float* bv = (const float*)d_in[7];
  const float* Wo = (const float*)d_in[8];
  const float* bo = (const float*)d_in[9];
  float* out = (float*)d_out;
  _Float16* Wh = (_Float16*)d_ws;   // 4*256*256 f16 = 512 KB scratch

  prep_weights_f16<<<1024, 256, 0, stream>>>(Wq, Wk, Wv, Wo, Wh);
  const int nGroups = B_ * T_;                     // 32768 (b,t) groups
  fused_rep_attn<<<nGroups / GROUPS_PER_WG, 256, 0, stream>>>(
      x, rm, Wh, bq, bk, bv, bo, out);
}
